// EdgeNodeOnlyModel_24558622999195
// MI455X (gfx1250) — matrix operand order repaired and checked by reading, not compiled, not run on hardware
//
#include <hip/hip_runtime.h>
#include <hip/hip_bf16.h>

typedef __attribute__((ext_vector_type(16))) _Float16 v16h;
typedef __attribute__((ext_vector_type(2)))  _Float16 h16x2;
typedef __attribute__((ext_vector_type(8)))  float    v8f;
typedef __attribute__((ext_vector_type(4)))  float    f32x4;
typedef __attribute__((ext_vector_type(2)))  float    f32x2;
typedef __attribute__((ext_vector_type(4)))  unsigned int u32x4;

#define LEAK 0.1f
#define EPSV 1e-5f
#define STATS_BLOCKS 256
#define ACT_STRIDE 72   // 64 features padded to 72 halves (144B row) -> conflict-free banks
#define NWAVES 8

// ---------------- deterministic BN statistics ----------------

__global__ __launch_bounds__(256) void e_stats_partial(const float* __restrict__ e,
                                                       float* __restrict__ part, int E) {
    __shared__ float red[256 * 20];
    int tid = threadIdx.x;
    float s[10], q[10];
#pragma unroll
    for (int c = 0; c < 10; ++c) { s[c] = 0.f; q[c] = 0.f; }
    for (int r = blockIdx.x * 256 + tid; r < E; r += 256 * STATS_BLOCKS) {
        const f32x2* p = (const f32x2*)(e + (size_t)r * 10);
#pragma unroll
        for (int c = 0; c < 5; ++c) {
            f32x2 v = p[c];
            s[2*c]   += v[0]; q[2*c]   += v[0] * v[0];
            s[2*c+1] += v[1]; q[2*c+1] += v[1] * v[1];
        }
    }
#pragma unroll
    for (int c = 0; c < 10; ++c) { red[tid*20 + c] = s[c]; red[tid*20 + 10 + c] = q[c]; }
    __syncthreads();
    if (tid < 20) {
        float a = 0.f;
        for (int i = 0; i < 256; ++i) a += red[i*20 + tid];
        part[blockIdx.x * 20 + tid] = a;
    }
}

__global__ __launch_bounds__(256) void x_stats_partial(const float* __restrict__ x,
                                                       float* __restrict__ part, int N) {
    __shared__ float red[256 * 32];
    int tid = threadIdx.x;
    float s[16], q[16];
#pragma unroll
    for (int c = 0; c < 16; ++c) { s[c] = 0.f; q[c] = 0.f; }
    for (int r = blockIdx.x * 256 + tid; r < N; r += 256 * STATS_BLOCKS) {
        const f32x4* p = (const f32x4*)(x + (size_t)r * 16);
#pragma unroll
        for (int w = 0; w < 4; ++w) {
            f32x4 v = p[w];
#pragma unroll
            for (int i = 0; i < 4; ++i) { float t = v[i]; s[w*4+i] += t; q[w*4+i] += t * t; }
        }
    }
#pragma unroll
    for (int c = 0; c < 16; ++c) { red[tid*32 + c] = s[c]; red[tid*32 + 16 + c] = q[c]; }
    __syncthreads();
    if (tid < 32) {
        float a = 0.f;
        for (int i = 0; i < 256; ++i) a += red[i*32 + tid];
        part[blockIdx.x * 32 + tid] = a;
    }
}

// sArr/tArr: per-feature affine xhat = raw*s + t for the 42 concat features
__global__ void bn_finalize(const float* __restrict__ epart, const float* __restrict__ xpart,
                            const float* __restrict__ gn, const float* __restrict__ bnb,
                            const float* __restrict__ ge, const float* __restrict__ be,
                            float* __restrict__ sArr, float* __restrict__ tArr, int N, int E) {
    int c = threadIdx.x;
    if (c < 16) {
        float s = 0.f, q = 0.f;
        for (int b = 0; b < STATS_BLOCKS; ++b) { s += xpart[b*32 + c]; q += xpart[b*32 + 16 + c]; }
        float invN = 1.f / (float)N;
        float mean = s * invN, var = q * invN - mean * mean;
        float sc = gn[c] * rsqrtf(var + EPSV);
        float sh = bnb[c] - mean * sc;
        sArr[c] = sc; tArr[c] = sh;
        sArr[c + 16] = sc; tArr[c + 16] = sh;   // dst features reuse node BN
    }
    if (c < 10) {
        float s = 0.f, q = 0.f;
        for (int b = 0; b < STATS_BLOCKS; ++b) { s += epart[b*20 + c]; q += epart[b*20 + 10 + c]; }
        float invE = 1.f / (float)E;
        float mean = s * invE, var = q * invE - mean * mean;
        float sc = ge[c] * rsqrtf(var + EPSV);
        float sh = be[c] - mean * sc;
        sArr[32 + c] = sc; tArr[32 + c] = sh;
    }
}

// ---------------- weight fragment prep (WMMA A layout, f16) ----------------
// frag f: 512 halves, index = lane*16 + j. 16x32 A in this code's contiguous k order:
//   row m = lane&15, K = j + 16*(lane>>4)
// frags 0..7: L1 (mt*2+kt), BN folded, bias at k==42
// frags 8..15: L2 ; 16..19: L3 ; 20: L4 ; 21: L5 (bias at k==16)
__global__ __launch_bounds__(256) void prep_frags(const float* __restrict__ W1, const float* __restrict__ b1,
                                                  const float* __restrict__ W2, const float* __restrict__ W3,
                                                  const float* __restrict__ W4, const float* __restrict__ W5,
                                                  const float* __restrict__ b5,
                                                  const float* __restrict__ sArr, const float* __restrict__ tArr,
                                                  _Float16* __restrict__ frags) {
    for (int i = threadIdx.x; i < 22 * 512; i += 256) {
        int f = i >> 9, r = i & 511, lane = r >> 4, j = r & 15;
        int m = lane & 15, h = lane >> 4;
        int K = j + 16 * h;
        float val = 0.f;
        if (f < 8) {
            int mt = f >> 1, kt = f & 1;
            int o = mt * 16 + m, k = kt * 32 + K;
            if (k < 42) val = sArr[k] * W1[k * 64 + o];
            else if (k == 42) {
                float a = b1[o];
                for (int kk = 0; kk < 42; ++kk) a += tArr[kk] * W1[kk * 64 + o];
                val = a;
            }
        } else if (f < 16) {
            int g = f - 8;
            int o = (g >> 1) * 16 + m, k = (g & 1) * 32 + K;
            val = W2[k * 64 + o];
        } else if (f < 20) {
            int g = f - 16;
            int o = (g >> 1) * 16 + m, k = (g & 1) * 32 + K;
            val = W3[k * 32 + o];
        } else if (f == 20) {
            val = W4[K * 16 + m];
        } else {
            int o = m, k = K;
            if (o < 2) {
                if (k < 16) val = W5[k * 2 + o];
                else if (k == 16) val = b5[o];
            }
        }
        frags[i] = (_Float16)val;
    }
}

// ---------------- fused edge-MLP (WMMA) ----------------

__device__ __forceinline__ v8f wmma_f16(v16h a, v16h b, v8f c) {
    return __builtin_amdgcn_wmma_f32_16x16x32_f16(false, a, false, b, (short)0, c, false, false);
}

__device__ __forceinline__ v8f bias8(const float* p) {
    const f32x4* q = (const f32x4*)p;
    f32x4 a = q[0], b = q[1];
    v8f c;
#pragma unroll
    for (int i = 0; i < 4; ++i) { c[i] = a[i]; c[4 + i] = b[i]; }
    return c;
}

// B f16 32x16, contiguous k order used here (matches A frags): lane l -> column n=l&15, half j -> K = 16*(l>>4) + j
__device__ __forceinline__ v16h loadB(const _Float16* row, int kt, int h) {
    union { v16h v; u32x4 q[2]; } u;
    const u32x4* p = (const u32x4*)(row + kt * 32 + 16 * h);
    u.q[0] = p[0]; u.q[1] = p[1];
    return u.v;
}

// D tile (f32) -> leaky-relu in packed f16 -> halves at features t*16+8h+{0..7}.
// lrelu(x) = max(x, LEAK*x) since 0 < LEAK < 1; done as v_pk_mul_f16 + v_pk_max
// after v_cvt_pk_f16_f32 packing (3 VALU per 2 elements, no cmp/cndmask chain).
__device__ __forceinline__ void storeTileLrelu(_Float16* row, int t, int h, v8f c) {
    h16x2 lk; lk[0] = (_Float16)LEAK; lk[1] = (_Float16)LEAK;
    union { h16x2 h2[4]; u32x4 q; } u;
#pragma unroll
    for (int i = 0; i < 4; ++i) {
        h16x2 p; p[0] = (_Float16)c[2*i]; p[1] = (_Float16)c[2*i+1];
        u.h2[i] = __builtin_elementwise_max(p, (h16x2)(p * lk));
    }
    *(u32x4*)(row + t * 16 + 8 * h) = u.q;
}

__global__ __launch_bounds__(256) void edge_mlp_wmma(const float* __restrict__ x,
                                                     const int* __restrict__ ei,
                                                     const float* __restrict__ e,
                                                     const _Float16* __restrict__ frags,
                                                     const float* __restrict__ b2,
                                                     const float* __restrict__ b3,
                                                     const float* __restrict__ b4,
                                                     float* __restrict__ out,
                                                     int nTiles, int E) {
    __shared__ __align__(16) _Float16 act[NWAVES * 16 * ACT_STRIDE];
    __shared__ __align__(16) float bias2[64];
    __shared__ __align__(16) float bias3[32];
    __shared__ __align__(16) float bias4[16];

    int tid = threadIdx.x;
    if (tid < 64) bias2[tid] = b2[tid];
    if (tid < 32) bias3[tid] = b3[tid];
    if (tid < 16) bias4[tid] = b4[tid];
    for (int i = tid; i < NWAVES * 16 * ACT_STRIDE; i += 256) act[i] = (_Float16)0.f;
    __syncthreads();

    int wv = tid >> 5, lane = tid & 31;
    int m = lane & 15, h = lane >> 4;
    _Float16* row = &act[wv * 16 * ACT_STRIDE + m * ACT_STRIDE];

    // loop-invariant weight A-fragments held in VGPRs
    v16h A[22];
#pragma unroll
    for (int f = 0; f < 22; ++f) {
        union { v16h v; u32x4 q[2]; } u;
        const u32x4* p = (const u32x4*)(frags + f * 512 + lane * 16);
        u.q[0] = p[0]; u.q[1] = p[1];
        A[f] = u.v;
    }

    int waveGlobal = blockIdx.x * NWAVES + wv;
    int nW = gridDim.x * NWAVES;

    for (int tile = waveGlobal; tile < nTiles; tile += nW) {
        int eIdx = tile * 16 + m;
        // B0: K = 16h+j -> h=0: src feats 0..15 ; h=1: dst feats 0..15
        int idx = ei[(size_t)h * (size_t)E + (size_t)eIdx];
        const f32x4* xp = (const f32x4*)(x + (size_t)idx * 16);
        f32x4 x0 = xp[0], x1 = xp[1], x2 = xp[2], x3 = xp[3];
        v16h B0;
#pragma unroll
        for (int i = 0; i < 4; ++i) {
            B0[i]      = (_Float16)x0[i];
            B0[4 + i]  = (_Float16)x1[i];
            B0[8 + i]  = (_Float16)x2[i];
            B0[12 + i] = (_Float16)x3[i];
        }
        // B1: K=32+16h+j -> h=0: e[0..9], 1.0 (bias col k=42), zeros ; h=1: zeros
        v16h B1;
#pragma unroll
        for (int i = 0; i < 16; ++i) B1[i] = (_Float16)0.f;
        if (h == 0) {
            const f32x2* epv = (const f32x2*)(e + (size_t)eIdx * 10);
#pragma unroll
            for (int i = 0; i < 5; ++i) {
                f32x2 v = epv[i];
                B1[2*i] = (_Float16)v[0];
                B1[2*i+1] = (_Float16)v[1];
            }
            B1[10] = (_Float16)1.0f;
        }

        // ---- L1: 64x64pad @ 64x16 ----
#pragma unroll
        for (int mt = 0; mt < 4; ++mt) {
            v8f c;
#pragma unroll
            for (int i = 0; i < 8; ++i) c[i] = 0.f;
            c = wmma_f16(A[2*mt + 0], B0, c);
            c = wmma_f16(A[2*mt + 1], B1, c);
            storeTileLrelu(row, mt, h, c);
        }
        asm volatile("s_wait_dscnt 0" ::: "memory");

        // ---- L2: 64x64 ----
        {
            v16h bb0 = loadB(row, 0, h);
            v16h bb1 = loadB(row, 1, h);
#pragma unroll
            for (int mt = 0; mt < 4; ++mt) {
                v8f c = bias8(&bias2[mt * 16 + 8 * h]);
                c = wmma_f16(A[8 + 2*mt + 0], bb0, c);
                c = wmma_f16(A[8 + 2*mt + 1], bb1, c);
                storeTileLrelu(row, mt, h, c);
            }
        }
        asm volatile("s_wait_dscnt 0" ::: "memory");

        // ---- L3: 32x64 ----
        {
            v16h bb0 = loadB(row, 0, h);
            v16h bb1 = loadB(row, 1, h);
#pragma unroll
            for (int mt = 0; mt < 2; ++mt) {
                v8f c = bias8(&bias3[mt * 16 + 8 * h]);
                c = wmma_f16(A[16 + 2*mt + 0], bb0, c);
                c = wmma_f16(A[16 + 2*mt + 1], bb1, c);
                storeTileLrelu(row, mt, h, c);
            }
        }
        asm volatile("s_wait_dscnt 0" ::: "memory");

        // ---- L4: 16x32 ----
        {
            v16h bb0 = loadB(row, 0, h);
            v8f c = bias8(&bias4[8 * h]);
            c = wmma_f16(A[20], bb0, c);
            storeTileLrelu(row, 0, h, c);
            if (h) { // constant-1 column for L5 bias (feature 16), zeros 17..23
                union { h16x2 h2; unsigned u; } one;
                one.h2[0] = (_Float16)1.0f; one.h2[1] = (_Float16)0.f;
                u32x4 q = { one.u, 0u, 0u, 0u };
                *(u32x4*)(row + 16) = q;
            }
        }
        asm volatile("s_wait_dscnt 0" ::: "memory");

        // ---- L5: 2(pad16) x 16(pad32) ----
        {
            v16h bb0 = loadB(row, 0, h);
            v8f c;
#pragma unroll
            for (int i = 0; i < 8; ++i) c[i] = 0.f;
            c = wmma_f16(A[21], bb0, c);
            if (h == 0) { // rows 0,1 of D live in lanes 0..15, c[0], c[1]
                f32x2 o; o[0] = c[0]; o[1] = c[1];
                *(f32x2*)(out + (size_t)eIdx * 2) = o;
            }
        }
    }
}

// ---------------- launch ----------------

extern "C" void kernel_launch(void* const* d_in, const int* in_sizes, int n_in,
                              void* d_out, int out_size, void* d_ws, size_t ws_size,
                              hipStream_t stream) {
    const float* x   = (const float*)d_in[0];
    const int*   ei  = (const int*)d_in[1];
    const float* e   = (const float*)d_in[2];
    // d_in[3] = xbatch (unused by the reference computation)
    const float* gn  = (const float*)d_in[4];
    const float* bnb = (const float*)d_in[5];
    const float* ge  = (const float*)d_in[6];
    const float* be  = (const float*)d_in[7];
    const float* W1 = (const float*)d_in[8];  const float* b1 = (const float*)d_in[9];
    const float* W2 = (const float*)d_in[10]; const float* b2 = (const float*)d_in[11];
    const float* W3 = (const float*)d_in[12]; const float* b3 = (const float*)d_in[13];
    const float* W4 = (const float*)d_in[14]; const float* b4 = (const float*)d_in[15];
    const float* W5 = (const float*)d_in[16]; const float* b5 = (const float*)d_in[17];

    int N = in_sizes[0] / 16;
    int E = in_sizes[2] / 10;

    char* ws = (char*)d_ws;
    float*    epart = (float*)(ws);              // 256*20 f32
    float*    xpart = (float*)(ws + 20480);      // 256*32 f32
    float*    sArr  = (float*)(ws + 53248);      // 64 f32
    float*    tArr  = (float*)(ws + 53504);      // 64 f32
    _Float16* frags = (_Float16*)(ws + 57344);   // 22 * 512 f16

    e_stats_partial<<<STATS_BLOCKS, 256, 0, stream>>>(e, epart, E);
    x_stats_partial<<<STATS_BLOCKS, 256, 0, stream>>>(x, xpart, N);
    bn_finalize<<<1, 32, 0, stream>>>(epart, xpart, gn, bnb, ge, be, sArr, tArr, N, E);
    prep_frags<<<1, 256, 0, stream>>>(W1, b1, W2, W3, W4, W5, b5, sArr, tArr, frags);

    int nTiles = E / 16;
    edge_mlp_wmma<<<2048, 256, 0, stream>>>(x, ei, e, frags, b2, b3, b4,
                                            (float*)d_out, nTiles, E);
}
